// SNet_texture_ver2_14783277433071
// MI455X (gfx1250) — compile-verified, matrix operand order checked by reading, not run on hardware
//
#include <hip/hip_runtime.h>

typedef __attribute__((ext_vector_type(16))) _Float16 v16h;
typedef __attribute__((ext_vector_type(8)))  _Float16 v8h;
typedef __attribute__((ext_vector_type(8)))  float    v8f;

#define TM 8                 // pixel rows per block tile (one wave per row)
#define TN 16                // pixel cols, small-conv tile
#define TNF 32               // pixel cols, fast-conv tile (2 M-tiles per wave)
#define PAD_H (TM + 2)
#define PAD_W (TN + 2)        // 18
#define PADWF (TNF + 2)       // 34
#define NPIX  (PAD_H * PAD_W)   // 180
#define NPIXF (PAD_H * PADWF)   // 340

// ---------------------------------------------------------------------------
// Weight prep -> WMMA B-fragment-major f16, tap-major K ordering:
//   GEMM k = tap*Cin_s + c,  fragment idx = ((chunk*NT+nt)*32+lane)*16+e,
//   kloc(e,lane) = e + (e>=8?8:0) + (lane>=16?8:0).
// ---------------------------------------------------------------------------
__global__ void prep_wfrag(const float* __restrict__ w, _Float16* __restrict__ wk,
                           int Cin, int Cin_s, int Cout, int nChunks, int NT)
{
    int idx = blockIdx.x * blockDim.x + threadIdx.x;
    int total = nChunks * 32 * NT * 16;
    if (idx >= total) return;
    int e    = idx & 15;
    int lane = (idx >> 4) & 31;
    int rest = idx >> 9;              // chunk*NT + nt
    int nt   = rest % NT;
    int c0   = rest / NT;
    int k = (c0 << 5) + e + ((e >= 8) ? 8 : 0) + ((lane >= 16) ? 8 : 0);
    int n = nt * 16 + (lane & 15);
    int t = k / Cin_s;                // tap index
    int c = k % Cin_s;                // channel index
    float v = 0.f;
    if (t < 9 && c < Cin && n < Cout)
        v = w[((size_t)n * Cin + c) * 9 + t];   // OIHW
    wk[idx] = (_Float16)v;
}

__device__ __forceinline__ v16h hcat(v8h a, v8h b) {
    return __builtin_shufflevector(a, b, 0,1,2,3,4,5,6,7,8,9,10,11,12,13,14,15);
}
__device__ __forceinline__ v8f wmma(v16h a, v16h b, v8f c) {
    return __builtin_amdgcn_wmma_f32_16x16x32_f16(false, a, false, b,
                                                  (short)0, c, false, false);
}

// ---------------------------------------------------------------------------
// Fast path: Cin = 64, NHWC. 8x32 pixel tile, 2 M-tiles per wave. 18 K-chunks,
// tap compile-time per chunk, every fragment = two contiguous ds_load_b128.
// All spatial dims divide the tile exactly (512/256/128/64), so no bounds
// guards in the epilogue.
// ---------------------------------------------------------------------------
template<int NT, bool RELU, bool OUTF32>
__global__ __launch_bounds__(256) void conv3x3_c64(
    const _Float16* __restrict__ in,      // [z][H][W][64]
    const _Float16* __restrict__ wfrag,   // 18*NT*512 halfs, fragment-major
    const float* __restrict__ bias,
    _Float16* __restrict__ out16,         // [z][H][W][CoutPad]
    float* __restrict__ out32,            // [z][H][W] (Cout==1 path)
    int H, int W, int Cout, int CoutPad)
{
    extern __shared__ _Float16 smem[];
    _Float16* sW = smem;                       // 576*NT*16
    _Float16* sA = smem + 576 * NT * 16;       // NPIXF*64, layout [pix][c]

    const int tid  = threadIdx.x;
    const int lane = tid & 31;
    const int wave = tid >> 5;
    const int x0   = blockIdx.x * TNF;
    const int y0   = blockIdx.y * TM;
    const int z    = blockIdx.z;
    const _Float16* inB = in + (size_t)z * H * W * 64;

    for (int i = tid; i < 576 * NT * 2; i += 256)
        *(v8h*)(sW + (size_t)i * 8) = *(const v8h*)(wfrag + (size_t)i * 8);

    for (int s = tid; s < NPIXF * 8; s += 256) {
        int pix = s >> 3, cg = (s & 7) << 3;
        int py = pix / PADWF, px = pix - py * PADWF;
        int gy = y0 + py - 1, gx = x0 + px - 1;
        v8h v = {};
        if (gy >= 0 && gy < H && gx >= 0 && gx < W)
            v = *(const v8h*)(inB + (((size_t)gy * W + gx) << 6) + cg);
        *(v8h*)(sA + pix * 64 + cg) = v;
    }
    __syncthreads();

    const int mrow = lane & 15;
    const int khi  = (lane >> 4) << 3;
    v8f acc0[NT] = {}, acc1[NT] = {};

    #pragma unroll
    for (int ck = 0; ck < 18; ++ck) {
        const int t  = ck >> 1;              // compile-time tap
        const int ty = t / 3, tx = t % 3;
        const int cb = (ck & 1) << 5;        // channel half 0/32
        const _Float16* ap0 = sA + ((wave + ty) * PADWF + (mrow + tx)) * 64 + cb + khi;
        const _Float16* ap1 = ap0 + 16 * 64; // second M-tile (px + 16)
        v16h a0 = hcat(*(const v8h*)ap0, *(const v8h*)(ap0 + 16));
        v16h a1 = hcat(*(const v8h*)ap1, *(const v8h*)(ap1 + 16));
        const _Float16* bp = sW + (size_t)(ck * NT) * 512 + lane * 16;
        #pragma unroll
        for (int nt = 0; nt < NT; ++nt) {
            v16h b = *(const v16h*)(bp + nt * 512);
            acc0[nt] = wmma(a0, b, acc0[nt]);
            acc1[nt] = wmma(a1, b, acc1[nt]);
        }
    }

    const int gy = y0 + wave;
    #pragma unroll
    for (int mt = 0; mt < 2; ++mt) {
        #pragma unroll
        for (int nt = 0; nt < NT; ++nt) {
            int n = nt * 16 + (lane & 15);
            float bval = (n < Cout) ? bias[n] : 0.f;
            #pragma unroll
            for (int r = 0; r < 8; ++r) {
                int gx = x0 + mt * 16 + r + khi;
                size_t pix = ((size_t)z * H + gy) * W + gx;
                float val = (mt ? acc1[nt][r] : acc0[nt][r]) + bval;
                if (RELU) val = fmaxf(val, 0.f);
                if (OUTF32) { if (n == 0) out32[pix] = val; }
                else out16[pix * CoutPad + n] = (_Float16)val;
            }
        }
    }
}

// ---------------------------------------------------------------------------
// Small path: Cin padded to 8, NHWC. K = 72 -> 3 chunks.  Each fragment half
// (8 elems) is ONE tap's 8 channels: taps t = 4ck + hodd and t+2.  The tap
// select is a branch-free integer mad (off = o0 + hodd*(o1-o0)); K-tail taps
// (>=9) statically map to the zero slot.
// ---------------------------------------------------------------------------
#define ZSLOT (NPIX * 8)

template<int NT, bool RELU>
__global__ __launch_bounds__(256) void conv3x3_c8(
    const _Float16* __restrict__ in,      // [z][H][W][8]
    const _Float16* __restrict__ wfrag,   // 3*NT*512 halfs
    const float* __restrict__ bias,
    _Float16* __restrict__ out16,         // [z][H][W][CoutPad]
    int H, int W, int Cout, int CoutPad)
{
    extern __shared__ _Float16 smem[];
    _Float16* sW = smem;                   // 96*NT*16
    _Float16* sA = smem + 96 * NT * 16;    // NPIX*8 + zero slot

    const int tid  = threadIdx.x;
    const int lane = tid & 31;
    const int wave = tid >> 5;
    const int x0   = blockIdx.x * TN;
    const int y0   = blockIdx.y * TM;
    const int z    = blockIdx.z;
    const _Float16* inB = in + (size_t)z * H * W * 8;

    for (int i = tid; i < 96 * NT * 2; i += 256)
        *(v8h*)(sW + (size_t)i * 8) = *(const v8h*)(wfrag + (size_t)i * 8);
    for (int pix = tid; pix < NPIX; pix += 256) {
        int py = pix / PAD_W, px = pix - py * PAD_W;
        int gy = y0 + py - 1, gx = x0 + px - 1;
        v8h v = {};
        if (gy >= 0 && gy < H && gx >= 0 && gx < W)
            v = *(const v8h*)(inB + (((size_t)gy * W + gx) << 3));
        *(v8h*)(sA + pix * 8) = v;
    }
    if (tid == 0) { v8h zz = {}; *(v8h*)(sA + ZSLOT) = zz; }
    __syncthreads();

    const int mrow = lane & 15;
    const int khi  = (lane >> 4) << 3;
    const int hodd = lane >> 4;            // 0/1: which tap parity this half-wave owns
    v8f acc[NT] = {};

    #pragma unroll
    for (int ck = 0; ck < 3; ++ck) {
        // compile-time-shaped offsets for taps 4ck .. 4ck+3 (>=9 -> zero slot)
        auto aoff = [&](int t) -> int {
            return (t <= 8) ? (((wave + t / 3) * PAD_W + (mrow + t % 3)) * 8) : ZSLOT;
        };
        int o0e = aoff(4 * ck),     o0o = aoff(4 * ck + 1);
        int o1e = aoff(4 * ck + 2), o1o = aoff(4 * ck + 3);
        int off0 = o0e + hodd * (o0o - o0e);     // branch-free select
        int off1 = o1e + hodd * (o1o - o1e);
        v8h a0 = *(const v8h*)(sA + off0);
        v8h a1 = *(const v8h*)(sA + off1);
        v16h a = hcat(a0, a1);
        const _Float16* bp = sW + (size_t)(ck * NT) * 512 + lane * 16;
        #pragma unroll
        for (int nt = 0; nt < NT; ++nt) {
            v16h b = *(const v16h*)(bp + nt * 512);
            acc[nt] = wmma(a, b, acc[nt]);
        }
    }

    const int gy = y0 + wave;
    #pragma unroll
    for (int nt = 0; nt < NT; ++nt) {
        int n = nt * 16 + (lane & 15);
        float bval = (n < Cout) ? bias[n] : 0.f;
        #pragma unroll
        for (int r = 0; r < 8; ++r) {
            int gx = x0 + r + khi;
            size_t pix = ((size_t)z * H + gy) * W + gx;
            float val = acc[nt][r] + bval;
            if (RELU) val = fmaxf(val, 0.f);
            out16[pix * CoutPad + n] = (_Float16)val;
        }
    }
}

// ---------------------------------------------------------------------------
// Nearest resize f32 -> f16 NHWC8 (channel 0 = data, 1..7 = 0). grid.y = batch.
// ---------------------------------------------------------------------------
__global__ void resize_f2h8(const float* __restrict__ s, _Float16* __restrict__ d,
                            int ih, int iw, int oh, int ow)
{
    int i = blockIdx.x * blockDim.x + threadIdx.x;
    int z = blockIdx.y;
    int tot = oh * ow;
    if (i >= tot) return;
    int y = i / ow, x = i % ow;
    int sy = y * ih / oh, sx = x * iw / ow;
    v8h v = {};
    v[0] = (_Float16)s[((size_t)z * ih + sy) * iw + sx];
    *(v8h*)(d + (((size_t)z * tot + i) << 3)) = v;
}

// ---------------------------------------------------------------------------
// Fused softmax(81) + 9x9 dynamic filter, single-pass online softmax.
// ---------------------------------------------------------------------------
__global__ void softmax_dynconv(const _Float16* __restrict__ logits,
                                const _Float16* __restrict__ xs,
                                float* __restrict__ ys, int H, int W)
{
    int i = blockIdx.x * blockDim.x + threadIdx.x;
    int z = blockIdx.y;
    int HW = H * W;
    if (i >= HW) return;
    int y = i / W, x = i % W;
    const _Float16* lg = logits + ((size_t)z * HW + i) * 96;
    const _Float16* xb = xs + (size_t)z * HW * 8;
    float m = -3.0e38f, s = 0.f, a = 0.f;
    int k = 0;
    for (int dy = -4; dy <= 4; ++dy) {
        int yy = y + dy;
        for (int dx = -4; dx <= 4; ++dx, ++k) {
            float l = (float)lg[k];
            int xx = x + dx;
            float xv = (yy >= 0 && yy < H && xx >= 0 && xx < W)
                           ? (float)xb[((size_t)yy * W + xx) << 3] : 0.f;
            float mn = fmaxf(m, l);
            float sc = __expf(m - mn);
            float e  = __expf(l - mn);
            s = s * sc + e;
            a = a * sc + e * xv;
            m = mn;
        }
    }
    ys[(size_t)z * HW + i] = a / s;
}

// ---------------------------------------------------------------------------
// Nearest upsample into concat buffer channel ch (NHWC8 f16).
// ---------------------------------------------------------------------------
__global__ void upsample_cat(const float* __restrict__ ys, _Float16* __restrict__ cat,
                             int sh, int sw, int ch, int OH, int OW)
{
    int i = blockIdx.x * blockDim.x + threadIdx.x;
    int z = blockIdx.y;
    int tot = OH * OW;
    if (i >= tot) return;
    int y = i / OW, x = i % OW;
    int sy = y * sh / OH, sx = x * sw / OW;
    cat[(((size_t)z * tot + i) << 3) + ch] =
        (_Float16)ys[((size_t)z * sh + sy) * sw + sx];
}

// ---------------------------------------------------------------------------

static inline unsigned cdiv(unsigned a, unsigned b) { return (a + b - 1) / b; }

extern "C" void kernel_launch(void* const* d_in, const int* in_sizes, int n_in,
                              void* d_out, int out_size, void* d_ws, size_t ws_size,
                              hipStream_t stream)
{
    const float* x   = (const float*)d_in[0];
    const float* g   = (const float*)d_in[1];
    const float* fWa = (const float*)d_in[2];   // (5,64,1,3,3)
    const float* fba = (const float*)d_in[3];
    const float* fWb = (const float*)d_in[4];   // (5,64,64,3,3)
    const float* fbb = (const float*)d_in[5];
    const float* fWc = (const float*)d_in[6];   // (5,81,64,3,3)
    const float* fbc = (const float*)d_in[7];
    const float* W2a = (const float*)d_in[8];   // (64,5,3,3)
    const float* b2a = (const float*)d_in[9];
    const float* W2b = (const float*)d_in[10];  // (1,64,3,3)
    const float* b2b = (const float*)d_in[11];
    float* out = (float*)d_out;
    (void)in_sizes; (void)n_in; (void)out_size; (void)ws_size;

    const int N = 2, H0 = 512, W0 = 512;
    const size_t P0 = (size_t)H0 * W0;

    char* ws = (char*)d_ws;
    size_t off = 0;
    auto alloc = [&](size_t bytes) -> void* {
        void* p = ws + off;
        off = (off + bytes + 255) & ~(size_t)255;
        return p;
    };
    const int shv[5] = {512, 256, 128, 64, 512};
    _Float16* xs[5];
    for (int i = 0; i < 5; ++i)
        xs[i] = (_Float16*)alloc((size_t)N * shv[i] * shv[i] * 8 * 2);
    _Float16* h1   = (_Float16*)alloc((size_t)N * P0 * 64 * 2);
    _Float16* h2   = (_Float16*)alloc((size_t)N * P0 * 64 * 2);
    _Float16* lgt  = (_Float16*)alloc((size_t)N * P0 * 96 * 2);
    float*    ysb  = (float*)   alloc((size_t)N * P0 * 4);
    _Float16* cat  = (_Float16*)alloc((size_t)N * P0 * 8 * 2);
    _Float16* wkS  = (_Float16*)alloc((size_t)3  * 4 * 512 * 2);
    _Float16* wkB  = (_Float16*)alloc((size_t)18 * 4 * 512 * 2);
    _Float16* wkC  = (_Float16*)alloc((size_t)18 * 6 * 512 * 2);
    _Float16* wk2b = (_Float16*)alloc((size_t)18 * 1 * 512 * 2);

    (void)hipMemsetAsync(cat, 0, (size_t)N * P0 * 8 * 2, stream);

    for (int i = 0; i < 5; ++i) {
        const float* src = (i == 4) ? g : x;
        int oh = shv[i];
        resize_f2h8<<<dim3(cdiv(oh * oh, 256), N), 256, 0, stream>>>(
            src, xs[i], H0, W0, oh, oh);
    }

    const size_t smC8    = (size_t)(96 * 4 * 16 + NPIX * 8 + 8) * 2;
    const size_t smC64n4 = (size_t)(576 * 4 * 16 + NPIXF * 64) * 2;
    const size_t smC64n6 = (size_t)(576 * 6 * 16 + NPIXF * 64) * 2;
    const size_t smC64n1 = (size_t)(576 * 1 * 16 + NPIXF * 64) * 2;

    for (int i = 0; i < 5; ++i) {
        int Hs = shv[i], Ws = shv[i];

        prep_wfrag<<<cdiv(3 * 4 * 512, 256), 256, 0, stream>>>(
            fWa + (size_t)i * 64 * 9, wkS, 1, 8, 64, 3, 4);
        prep_wfrag<<<cdiv(18 * 4 * 512, 256), 256, 0, stream>>>(
            fWb + (size_t)i * 64 * 576, wkB, 64, 64, 64, 18, 4);
        prep_wfrag<<<cdiv(18 * 6 * 512, 256), 256, 0, stream>>>(
            fWc + (size_t)i * 81 * 576, wkC, 64, 64, 81, 18, 6);

        dim3 cg8(cdiv(Ws, TN), cdiv(Hs, TM), N);
        dim3 cg64(cdiv(Ws, TNF), cdiv(Hs, TM), N);
        conv3x3_c8<4, true><<<cg8, 256, smC8, stream>>>(
            xs[i], wkS, fba + i * 64, h1, Hs, Ws, 64, 64);
        conv3x3_c64<4, true, false><<<cg64, 256, smC64n4, stream>>>(
            h1, wkB, fbb + i * 64, h2, nullptr, Hs, Ws, 64, 64);
        conv3x3_c64<6, false, false><<<cg64, 256, smC64n6, stream>>>(
            h2, wkC, fbc + i * 81, lgt, nullptr, Hs, Ws, 81, 96);

        softmax_dynconv<<<dim3(cdiv(Hs * Ws, 256), N), 256, 0, stream>>>(
            lgt, xs[i], ysb, Hs, Ws);
        upsample_cat<<<dim3(cdiv(H0 * W0, 256), N), 256, 0, stream>>>(
            ysb, cat, Hs, Ws, i, H0, W0);
    }

    prep_wfrag<<<cdiv(3 * 4 * 512, 256), 256, 0, stream>>>(
        W2a, wkS, 5, 8, 64, 3, 4);
    dim3 fg8(cdiv(W0, TN), cdiv(H0, TM), N);
    dim3 fg64(cdiv(W0, TNF), cdiv(H0, TM), N);
    conv3x3_c8<4, true><<<fg8, 256, smC8, stream>>>(
        cat, wkS, b2a, h1, H0, W0, 64, 64);

    prep_wfrag<<<cdiv(18 * 1 * 512, 256), 256, 0, stream>>>(
        W2b, wk2b, 64, 64, 1, 18, 1);
    conv3x3_c64<1, false, true><<<fg64, 256, smC64n1, stream>>>(
        h1, wk2b, b2b, nullptr, out, H0, W0, 1, 1);
}
